// NeuralGraph_89859305766987
// MI455X (gfx1250) — matrix operand order repaired and checked by reading, not compiled, not run on hardware
//
#include <hip/hip_runtime.h>
#include <hip/hip_fp16.h>
#include <hip/hip_bf16.h>

#define BN 4
#define NN 512
#define CHN 8
#define DMSG 24

typedef __attribute__((ext_vector_type(16))) _Float16 v16h;
typedef __attribute__((ext_vector_type(8)))  _Float16 v8h;
typedef __attribute__((ext_vector_type(8)))  float    v8f;

// Fast SiLU: denominator 1+exp(-x) >= 1, so hardware v_rcp_f32 (1 ulp) is safe.
__device__ __forceinline__ float silu_f(float x) {
    return x * __builtin_amdgcn_rcpf(1.0f + __expf(-x));
}

// ---------------------------------------------------------------------------
// Kernel 1: attention MLP per node: [B*N,8] -> 64 -> 32 -> 32 (fk,fq,bk,bq)
// ---------------------------------------------------------------------------
__global__ void att_mlp_kernel(const float* __restrict__ nodes,
                               const float* __restrict__ w1, const float* __restrict__ b1,
                               const float* __restrict__ w2, const float* __restrict__ b2,
                               const float* __restrict__ w3, const float* __restrict__ b3,
                               float* __restrict__ fk, float* __restrict__ fq,
                               float* __restrict__ bk, float* __restrict__ bq) {
    int t = blockIdx.x * blockDim.x + threadIdx.x;
    if (t >= BN * NN) return;
    float x[8];
#pragma unroll
    for (int c = 0; c < 8; ++c) x[c] = nodes[(size_t)t * 8 + c];
    float h1[64];
    for (int o = 0; o < 64; ++o) {
        float a = b1[o];
#pragma unroll
        for (int k = 0; k < 8; ++k) a += x[k] * w1[k * 64 + o];
        h1[o] = silu_f(a);
    }
    float h2[32];
    for (int o = 0; o < 32; ++o) {
        float a = b2[o];
        for (int k = 0; k < 64; ++k) a += h1[k] * w2[k * 32 + o];
        h2[o] = silu_f(a);
    }
    float ov[32];
    for (int o = 0; o < 32; ++o) {
        float a = b3[o];
        for (int k = 0; k < 32; ++k) a += h2[k] * w3[k * 32 + o];
        ov[o] = a;
    }
#pragma unroll
    for (int c = 0; c < 8; ++c) {
        fk[(size_t)t * 8 + c] = ov[c];
        fq[(size_t)t * 8 + c] = ov[8 + c];
        bk[(size_t)t * 8 + c] = ov[16 + c];
        bq[(size_t)t * 8 + c] = ov[24 + c];
    }
}

// ---------------------------------------------------------------------------
// Kernel 2 (hot): message MLP over all (b,i,j) pairs via WMMA f16->f32.
// One wave per workgroup; tile = 16 rows (fixed b,i; j0..j0+15).
// ---------------------------------------------------------------------------
__global__ __launch_bounds__(32)
void msg_mlp_kernel(const float* __restrict__ nodes, const float* __restrict__ edges,
                    const float* __restrict__ w1, const float* __restrict__ b1,
                    const float* __restrict__ w2, const float* __restrict__ b2,
                    const float* __restrict__ w3, const float* __restrict__ b3,
                    float* __restrict__ ma, float* __restrict__ mb,
                    float* __restrict__ out_edges) {
    __shared__ __align__(16) _Float16 sH1[16 * 72];   // hidden1 16x64, stride 72 halves
    __shared__ __align__(16) _Float16 sH2[16 * 40];   // hidden2 16x32, stride 40 halves
    __shared__ __align__(16) float    sM[16 * 24];    // m 16x24

    const int lane = threadIdx.x;
    const int l16  = lane & 15;
    const int hi   = lane >> 4;

    // ---- Preload weights into WMMA B-matrix register layout ----
    // B layout (16x16x32 f16): lane n=l16, K = (hi?8:0) + e + (e<8?0:8).
    v16h B1[4], B2[2][2], B3[2];
#pragma unroll
    for (int t = 0; t < 4; ++t) {
        int n  = 16 * t + l16;
        int kb = hi ? 8 : 0;
#pragma unroll
        for (int e = 0; e < 16; ++e) {
            int k = kb + e + (e < 8 ? 0 : 8);
            B1[t][e] = (k < DMSG) ? (_Float16)w1[k * 64 + n] : (_Float16)0.0f;
        }
    }
#pragma unroll
    for (int s = 0; s < 2; ++s)
#pragma unroll
        for (int t = 0; t < 2; ++t) {
            int n  = 16 * t + l16;
            int kb = 32 * s + (hi ? 8 : 0);
#pragma unroll
            for (int e = 0; e < 16; ++e)
                B2[s][t][e] = (_Float16)w2[(kb + e + (e < 8 ? 0 : 8)) * 32 + n];
        }
#pragma unroll
    for (int t = 0; t < 2; ++t) {
        int n  = 16 * t + l16;
        int kb = hi ? 8 : 0;
#pragma unroll
        for (int e = 0; e < 16; ++e)
            B3[t][e] = (n < DMSG) ? (_Float16)w3[(kb + e + (e < 8 ? 0 : 8)) * DMSG + n] : (_Float16)0.0f;
    }
    float bias1v[4], bias2v[2], bias3v[2];
#pragma unroll
    for (int t = 0; t < 4; ++t) bias1v[t] = b1[16 * t + l16];
#pragma unroll
    for (int t = 0; t < 2; ++t) bias2v[t] = b2[16 * t + l16];
#pragma unroll
    for (int t = 0; t < 2; ++t) bias3v[t] = (16 * t + l16 < DMSG) ? b3[16 * t + l16] : 0.0f;

    const int ntiles = BN * NN * (NN / 16);
    for (int tile = blockIdx.x; tile < ntiles; tile += gridDim.x) {
        const int jt = tile & 31;
        const int bi = tile >> 5;
        const int i  = bi & (NN - 1);
        const int b  = bi >> 9;
        const int j0 = jt * 16;

        // Prefetch next tile's edge block (512B) -> global_prefetch_b8
        {
            int tn = tile + gridDim.x;
            if (tn < ntiles && lane < 4) {
                int jt2 = tn & 31, bi2 = tn >> 5;
                int i2 = bi2 & (NN - 1), b2i = bi2 >> 9;
                size_t pb = (((size_t)(b2i * NN + i2)) * NN + jt2 * 16) * 8;
                __builtin_prefetch(edges + pb + (size_t)lane * 32, 0, 1);
            }
        }

        // ---- Build A1 (16x32 f16) in ISA VGPR layout ----
        // lanes 0-15: row M=l16 (j=j0+l16), K 0..7 = nodes[b,j], K 16..23 = edges[b,i,j]
        // lanes 16-31: row M=l16,            K 8..15 = nodes[b,i], K 24..31 = 0
        v16h A1;
        {
            int row = hi ? i : (j0 + l16);
            const float* np = nodes + ((size_t)(b * NN + row)) * 8;
#pragma unroll
            for (int e = 0; e < 8; ++e) A1[e] = (_Float16)np[e];
            if (!hi) {
                const float* ep = edges + (((size_t)(b * NN + i)) * NN + (j0 + l16)) * 8;
#pragma unroll
                for (int e = 0; e < 8; ++e) A1[8 + e] = (_Float16)ep[e];
            } else {
#pragma unroll
                for (int e = 0; e < 8; ++e) A1[8 + e] = (_Float16)0.0f;
            }
        }

        // ---- GEMM1: 16x24(pad32) @ 24x64 + bias ----
        v8f acc1[4];
#pragma unroll
        for (int t = 0; t < 4; ++t) {
            v8f c;
#pragma unroll
            for (int e = 0; e < 8; ++e) c[e] = bias1v[t];
            acc1[t] = __builtin_amdgcn_wmma_f32_16x16x32_f16(false, A1, false, B1[t],
                                                             (short)0, c, false, false);
        }
        // SiLU -> LDS (f16), stride 72
#pragma unroll
        for (int t = 0; t < 4; ++t) {
            int n = 16 * t + l16;
#pragma unroll
            for (int r = 0; r < 8; ++r)
                sH1[(r + 8 * hi) * 72 + n] = (_Float16)silu_f(acc1[t][r]);
        }
        asm volatile("s_wait_dscnt 0" ::: "memory");

        // ---- Load A2 (two 16x32 f16 A tiles, K split) ----
        v16h A2[2];
#pragma unroll
        for (int s = 0; s < 2; ++s) {
            const _Float16* p = sH1 + l16 * 72 + 32 * s + (hi ? 8 : 0);
            v8h lo = *(const v8h*)(p);
            v8h hh = *(const v8h*)(p + 16);
#pragma unroll
            for (int e = 0; e < 8; ++e) { A2[s][e] = lo[e]; A2[s][8 + e] = hh[e]; }
        }
        // ---- GEMM2: 16x64 @ 64x32 + bias ----
        v8f acc2[2];
#pragma unroll
        for (int t = 0; t < 2; ++t) {
            v8f c;
#pragma unroll
            for (int e = 0; e < 8; ++e) c[e] = bias2v[t];
            c = __builtin_amdgcn_wmma_f32_16x16x32_f16(false, A2[0], false, B2[0][t],
                                                       (short)0, c, false, false);
            c = __builtin_amdgcn_wmma_f32_16x16x32_f16(false, A2[1], false, B2[1][t],
                                                       (short)0, c, false, false);
            acc2[t] = c;
        }
        // SiLU -> LDS, stride 40
#pragma unroll
        for (int t = 0; t < 2; ++t) {
            int n = 16 * t + l16;
#pragma unroll
            for (int r = 0; r < 8; ++r)
                sH2[(r + 8 * hi) * 40 + n] = (_Float16)silu_f(acc2[t][r]);
        }
        asm volatile("s_wait_dscnt 0" ::: "memory");

        // ---- Load A3, GEMM3: 16x32 @ 32x24(pad32) + bias ----
        v16h A3;
        {
            const _Float16* p = sH2 + l16 * 40 + (hi ? 8 : 0);
            v8h lo = *(const v8h*)(p);
            v8h hh = *(const v8h*)(p + 16);
#pragma unroll
            for (int e = 0; e < 8; ++e) { A3[e] = lo[e]; A3[8 + e] = hh[e]; }
        }
        v8f acc3[2];
#pragma unroll
        for (int t = 0; t < 2; ++t) {
            v8f c;
#pragma unroll
            for (int e = 0; e < 8; ++e) c[e] = bias3v[t];
            acc3[t] = __builtin_amdgcn_wmma_f32_16x16x32_f16(false, A3, false, B3[t],
                                                             (short)0, c, false, false);
        }

        // ---- Transpose m through LDS, write coalesced ----
#pragma unroll
        for (int r = 0; r < 8; ++r)
            sM[(r + 8 * hi) * 24 + l16] = acc3[0][r];
        if (l16 < 8) {
#pragma unroll
            for (int r = 0; r < 8; ++r)
                sM[(r + 8 * hi) * 24 + 16 + l16] = acc3[1][r];
        }
        asm volatile("s_wait_dscnt 0" ::: "memory");

        {
            const size_t base = (((size_t)(b * NN + i)) * NN + j0) * 8;
            const int jj = lane >> 1;
            const int cc = (lane & 1) * 4;
            float4 va  = *(const float4*)&sM[jj * 24 + cc];
            float4 vb  = *(const float4*)&sM[jj * 24 + 8 + cc];
            float4 vab = *(const float4*)&sM[jj * 24 + 16 + cc];
            *(float4*)&ma[base + (size_t)lane * 4] = va;
            *(float4*)&mb[base + (size_t)lane * 4] = vb;
            float4 ev = *(const float4*)&edges[base + (size_t)lane * 4];
            float4 eo;
            eo.x = fminf(fmaxf(ev.x + vab.x, -100.0f), 100.0f);
            eo.y = fminf(fmaxf(ev.y + vab.y, -100.0f), 100.0f);
            eo.z = fminf(fmaxf(ev.z + vab.z, -100.0f), 100.0f);
            eo.w = fminf(fmaxf(ev.w + vab.w, -100.0f), 100.0f);
            *(float4*)&out_edges[base + (size_t)lane * 4] = eo;
        }
    }
}

// ---------------------------------------------------------------------------
// Kernel 3: online-softmax attention aggregation (one wave per output row).
// blocks [0, B*N)        -> agg_a over senders i   (softmax over i, fixed j)
// blocks [B*N, 2*B*N)    -> agg_b over receivers j (softmax over j, fixed i)
// ---------------------------------------------------------------------------
__global__ __launch_bounds__(32)
void agg_kernel(const float* __restrict__ fk, const float* __restrict__ fq,
                const float* __restrict__ bk, const float* __restrict__ bq,
                const float* __restrict__ ma, const float* __restrict__ mb,
                float* __restrict__ agga, float* __restrict__ aggb) {
    const float scale = 0.35355339059327373f;  // 1/sqrt(8)
    const int lane = threadIdx.x;
    const int blk  = blockIdx.x;
    const bool isA = blk < BN * NN;
    const int idx  = isA ? blk : blk - BN * NN;
    const int b = idx >> 9;
    const int r = idx & (NN - 1);  // j for fwd, i for bwd

    const float* qp = isA ? (fq + ((size_t)(b * NN + r)) * 8)
                          : (bq + ((size_t)(b * NN + r)) * 8);
    float q[8];
#pragma unroll
    for (int c = 0; c < 8; ++c) q[c] = qp[c];
    const float* kp = isA ? fk : bk;

    float mrun = -1e30f, srun = 0.0f, acc[8];
#pragma unroll
    for (int c = 0; c < 8; ++c) acc[c] = 0.0f;

    for (int o = lane; o < NN; o += 32) {
        const float* kr = kp + ((size_t)(b * NN + o)) * 8;
        float l = 0.0f;
#pragma unroll
        for (int c = 0; c < 8; ++c) l += q[c] * kr[c];
        l *= scale;
        const float* vp = isA ? (ma + (((size_t)(b * NN + o)) * NN + r) * 8)
                              : (mb + (((size_t)(b * NN + r)) * NN + o) * 8);
        float nm = fmaxf(mrun, l);
        float e0 = __expf(mrun - nm), e1 = __expf(l - nm);
        srun = srun * e0 + e1;
#pragma unroll
        for (int c = 0; c < 8; ++c) acc[c] = acc[c] * e0 + e1 * vp[c];
        mrun = nm;
    }
    // wave32 butterfly merge of (max, sum, acc[8])
#pragma unroll
    for (int off = 16; off >= 1; off >>= 1) {
        float m2 = __shfl_xor(mrun, off, 32);
        float s2 = __shfl_xor(srun, off, 32);
        float nm = fmaxf(mrun, m2);
        float e0 = __expf(mrun - nm), e1 = __expf(m2 - nm);
        srun = srun * e0 + s2 * e1;
#pragma unroll
        for (int c = 0; c < 8; ++c) {
            float a2 = __shfl_xor(acc[c], off, 32);
            acc[c] = acc[c] * e0 + a2 * e1;
        }
        mrun = nm;
    }
    if (lane == 0) {
        // srun >= 1 element contribution > 0; fast reciprocal is safe.
        float inv = __builtin_amdgcn_rcpf(srun);
        float* op = (isA ? agga : aggb) + ((size_t)(b * NN + r)) * 8;
#pragma unroll
        for (int c = 0; c < 8; ++c) op[c] = acc[c] * inv;
    }
}

// ---------------------------------------------------------------------------
// Kernel 4: out-projections + update MLP + clip -> new_nodes
// ---------------------------------------------------------------------------
__global__ void update_kernel(const float* __restrict__ nodes,
                              const float* __restrict__ agga, const float* __restrict__ aggb,
                              const float* __restrict__ outaw, const float* __restrict__ outbw,
                              const float* __restrict__ w1, const float* __restrict__ b1,
                              const float* __restrict__ w2, const float* __restrict__ b2,
                              const float* __restrict__ w3, const float* __restrict__ b3,
                              float* __restrict__ new_nodes) {
    int t = blockIdx.x * blockDim.x + threadIdx.x;
    if (t >= BN * NN) return;
    const float* ga = agga + (size_t)t * 8;
    const float* gb = aggb + (size_t)t * 8;
    float x[24];
#pragma unroll
    for (int c = 0; c < 8; ++c) {
        float sa = 0.0f, sb = 0.0f;
#pragma unroll
        for (int k = 0; k < 8; ++k) {
            sa += ga[k] * outaw[k * 8 + c];
            sb += gb[k] * outbw[k * 8 + c];
        }
        x[c] = sa; x[8 + c] = sb; x[16 + c] = nodes[(size_t)t * 8 + c];
    }
    float h1[64];
    for (int o = 0; o < 64; ++o) {
        float a = b1[o];
        for (int k = 0; k < 24; ++k) a += x[k] * w1[k * 64 + o];
        h1[o] = silu_f(a);
    }
    float h2[32];
    for (int o = 0; o < 32; ++o) {
        float a = b2[o];
        for (int k = 0; k < 64; ++k) a += h1[k] * w2[k * 32 + o];
        h2[o] = silu_f(a);
    }
#pragma unroll
    for (int c = 0; c < 8; ++c) {
        float u = b3[c];
        for (int k = 0; k < 32; ++k) u += h2[k] * w3[k * 8 + c];
        // DECAY=0, DT=1: nodes + upd, clipped
        float nv = nodes[(size_t)t * 8 + c] + u;
        new_nodes[(size_t)t * 8 + c] = fminf(fmaxf(nv, -100.0f), 100.0f);
    }
}

// ---------------------------------------------------------------------------
extern "C" void kernel_launch(void* const* d_in, const int* in_sizes, int n_in,
                              void* d_out, int out_size, void* d_ws, size_t ws_size,
                              hipStream_t stream) {
    const float* nodes  = (const float*)d_in[0];
    const float* edges  = (const float*)d_in[1];
    const float* msg_w1 = (const float*)d_in[2];
    const float* msg_b1 = (const float*)d_in[3];
    const float* msg_w2 = (const float*)d_in[4];
    const float* msg_b2 = (const float*)d_in[5];
    const float* msg_w3 = (const float*)d_in[6];
    const float* msg_b3 = (const float*)d_in[7];
    const float* upd_w1 = (const float*)d_in[8];
    const float* upd_b1 = (const float*)d_in[9];
    const float* upd_w2 = (const float*)d_in[10];
    const float* upd_b2 = (const float*)d_in[11];
    const float* upd_w3 = (const float*)d_in[12];
    const float* upd_b3 = (const float*)d_in[13];
    const float* att_w1 = (const float*)d_in[14];
    const float* att_b1 = (const float*)d_in[15];
    const float* att_w2 = (const float*)d_in[16];
    const float* att_b2 = (const float*)d_in[17];
    const float* att_w3 = (const float*)d_in[18];
    const float* att_b3 = (const float*)d_in[19];
    const float* outa_w = (const float*)d_in[20];
    const float* outb_w = (const float*)d_in[21];

    float* ws = (float*)d_ws;
    const size_t nnodes = (size_t)BN * NN * CHN;            // 16384
    const size_t npairs = (size_t)BN * NN * NN * CHN;       // 8388608
    float* fk   = ws;
    float* fq   = ws + nnodes;
    float* bk   = ws + 2 * nnodes;
    float* bq   = ws + 3 * nnodes;
    float* ma   = ws + 4 * nnodes;
    float* mbuf = ma + npairs;
    float* agga = mbuf + npairs;
    float* aggb = agga + nnodes;

    float* out_nodes = (float*)d_out;
    float* out_edges = out_nodes + nnodes;

    att_mlp_kernel<<<16, 128, 0, stream>>>(nodes, att_w1, att_b1, att_w2, att_b2,
                                           att_w3, att_b3, fk, fq, bk, bq);
    msg_mlp_kernel<<<8192, 32, 0, stream>>>(nodes, edges, msg_w1, msg_b1, msg_w2, msg_b2,
                                            msg_w3, msg_b3, ma, mbuf, out_edges);
    agg_kernel<<<2 * BN * NN, 32, 0, stream>>>(fk, fq, bk, bq, ma, mbuf, agga, aggb);
    update_kernel<<<8, 256, 0, stream>>>(nodes, agga, aggb, outa_w, outb_w,
                                         upd_w1, upd_b1, upd_w2, upd_b2, upd_w3, upd_b3,
                                         out_nodes);
}
